// BrainAgeGATv2_18554258719300
// MI455X (gfx1250) — matrix operand order repaired and checked by reading, not compiled, not run on hardware
//
#include <hip/hip_runtime.h>
#include <math.h>

// ---------------- types for WMMA (CDNA5 gfx1250, wave32) ----------------
typedef __attribute__((ext_vector_type(16))) __bf16 v16bf;
typedef __attribute__((ext_vector_type(8)))  float  v8f;
typedef __attribute__((ext_vector_type(4)))  unsigned int uint4v;

union BF16x16 { v16bf v; uint4v u[2]; };

__device__ __forceinline__ unsigned short f32_to_bf16(float f) {
  unsigned int u = __float_as_uint(f);
  unsigned int r = (u + 0x7FFFu + ((u >> 16) & 1u)) >> 16;  // RNE
  return (unsigned short)r;
}

__device__ __forceinline__ float lrelu(float v) { return v > 0.f ? v : 0.2f * v; }
__device__ __forceinline__ float relu_(float v) { return v > 0.f ? v : 0.f; }

// float atomic max via monotone int/uint punning (init must be -inf)
__device__ __forceinline__ void atomicMaxF(float* addr, float v) {
  if (v >= 0.0f) atomicMax((int*)addr, __float_as_int(v));
  else           atomicMin((unsigned int*)addr, (unsigned int)__float_as_int(v));
}

// ---------------- small utility kernels ----------------
__global__ void fillf_k(float* p, float v, int n) {
  int i = blockIdx.x * blockDim.x + threadIdx.x;
  if (i < n) p[i] = v;
}

// deg[d] += 1 ; la[d] += edge_attr[e]   (for fill_value='mean' self loops)
__global__ void deg_sum_k(const int* ei, const float* eattr, float* deg, float* la, int E) {
  int e = blockIdx.x * blockDim.x + threadIdx.x;
  if (e >= E) return;
  int d = ei[E + e];
  atomicAdd(&deg[d], 1.0f);
  atomicAdd(&la[d], eattr[e]);
}

__global__ void loop_div_k(float* la, const float* deg, int n) {
  int i = blockIdx.x * blockDim.x + threadIdx.x;
  if (i < n) la[i] = la[i] / fmaxf(deg[i], 1.0f);
}

// h[n, 0:64] = relu(x[n,0:4] @ W[4,64] + b), stored with row stride 128
__global__ void embed_k(const float* x, const float* W, const float* b, float* h, int n) {
  int t = blockIdx.x * blockDim.x + threadIdx.x;
  if (t >= n * 64) return;
  int nd = t >> 6, c = t & 63;
  const float* xp = x + (size_t)nd * 4;
  float acc = b[c];
#pragma unroll
  for (int i = 0; i < 4; ++i) acc += xp[i] * W[i * 64 + c];
  h[(size_t)nd * 128 + c] = relu_(acc);
}

// convert current activation (f32, stride 128, first `din` cols) to bf16 (stride 128)
__global__ void cvt_h_k(const float* h, unsigned short* hb, int n, int din) {
  int t = blockIdx.x * blockDim.x + threadIdx.x;
  if (t >= n * din) return;
  int nd = t / din, c = t - nd * din;
  size_t idx = (size_t)nd * 128 + c;
  hb[idx] = f32_to_bf16(h[idx]);
}

// wt[col][k] = bf16( col<128 ? Wl[k][col] : Wr[k][col-128] )  -> [256][din]
__global__ void prep_w_k(const float* Wl, const float* Wr, unsigned short* wt, int din) {
  int t = blockIdx.x * blockDim.x + threadIdx.x;
  if (t >= 256 * din) return;
  int col = t / din, k = t - col * din;
  float v = (col < 128) ? Wl[(size_t)k * 128 + col] : Wr[(size_t)k * 128 + (col - 128)];
  wt[(size_t)col * din + k] = f32_to_bf16(v);
}

// ---------------- WMMA GEMM: [N,din]x[din,256] -> xl[N,128], xr[N,128] ----------------
// block = 512 threads (16 waves), covers 64 rows x 256 cols.
// Weight panel [256][din] (<=64KB bf16) + 64-row A slab staged in LDS (padded pitch
// for bank spread); inner loop feeds v_wmma from ds_load_b128.
#define SW_PAD 8
#define SA_PITCH (128 + SW_PAD)
__global__ __launch_bounds__(512) void gemm_wmma_k(
    const unsigned short* __restrict__ hb,  // bf16 [N,128] (din valid)
    const unsigned short* __restrict__ wt,  // bf16 [256,din] (pre-transposed)
    const float* __restrict__ bl, const float* __restrict__ br,
    float* __restrict__ xl, float* __restrict__ xr,
    int n, int din) {
  __shared__ unsigned short sW[256 * (128 + SW_PAD)];  // <= 69.6 KB
  __shared__ unsigned short sA[64 * SA_PITCH];         // 17.4 KB

  const int tid = threadIdx.x;
  const int rowBase = blockIdx.x * 64;
  const int swPitch = din + SW_PAD;
  const int kChunks = din >> 3;                        // uint4 (8 bf16) chunks per col

  // stage weights: 32*din uint4 chunks (multiple of 512 for din in {64,128})
  {
    const int totalW = 32 * din;
    for (int i = tid; i < totalW; i += 512) {
      int col = i / kChunks, kp = i - col * kChunks;
      *(uint4v*)(sW + (size_t)col * swPitch + kp * 8) =
          *(const uint4v*)(wt + (size_t)col * din + kp * 8);
    }
  }
  // stage A slab: 64 rows x 128 cols = 1024 uint4 chunks (2 per thread)
  {
    for (int i = tid; i < 1024; i += 512) {
      int r = i >> 4, cp = i & 15;
      int gr = rowBase + r; if (gr >= n) gr = n - 1;   // clamp (EXEC stays full)
      *(uint4v*)(sA + (size_t)r * SA_PITCH + cp * 8) =
          *(const uint4v*)(hb + (size_t)gr * 128 + cp * 8);
    }
  }
  __syncthreads();

  const int wave = tid >> 5;
  const int lane = tid & 31;
  const int row16 = lane & 15;
  const int hi = lane >> 4;                 // 0: lanes 0-15, 1: lanes 16-31
  const int rowTile = wave & 3;             // which 16-row slab
  const int colBase = (wave >> 2) * 64;     // which 64-col slab of 256
  const int lrow = rowTile * 16 + row16;

  v8f acc[4] = {};
  for (int k = 0; k < din; k += 32) {
    // A fragment (16-bit A layout): lanes 0-15 K=k..k+7 / k+16..23, lanes 16-31 +8
    BF16x16 a;
    const unsigned short* ap = sA + (size_t)lrow * SA_PITCH + k + hi * 8;
    a.u[0] = *(const uint4v*)(ap);
    a.u[1] = *(const uint4v*)(ap + 16);
#pragma unroll
    for (int t = 0; t < 4; ++t) {
      const int col = colBase + t * 16 + row16;        // output column (0..255)
      // B fragment: lane = column; lanes 0-15 K=k..k+7 / k+16..23, lanes 16-31 +8
      BF16x16 b;
      const unsigned short* bp = sW + (size_t)col * swPitch + k + hi * 8;
      b.u[0] = *(const uint4v*)(bp);
      b.u[1] = *(const uint4v*)(bp + 16);
      acc[t] = __builtin_amdgcn_wmma_f32_16x16x32_bf16(
          false, a.v, false, b.v, (short)0, acc[t], false, false);
    }
  }
  // D layout: lane 0-15 -> M = j, lane 16-31 -> M = j+8 ; N = lane&15
#pragma unroll
  for (int t = 0; t < 4; ++t) {
    const int col = colBase + t * 16 + row16;
    const bool isL = col < 128;
    const int oc = isL ? col : col - 128;
    const float bias = isL ? bl[oc] : br[oc];
    float* outp = isL ? xl : xr;
#pragma unroll
    for (int j = 0; j < 8; ++j) {
      const int m = rowBase + rowTile * 16 + j + hi * 8;
      if (m < n) outp[(size_t)m * 128 + oc] = acc[t][j] + bias;
    }
  }
}

// ---------------- edge passes ----------------
// logits[e][h] = sum_c lrelu(xl[src]+xr[dst]+ea*We)[h,c] * att[h,c]; atomic max into mx[dst][h]
__global__ void edge_logits_k(const int* __restrict__ ei, const float* __restrict__ eattr,
                              const float* __restrict__ la,
                              const float* __restrict__ xl, const float* __restrict__ xr,
                              const float* __restrict__ We, const float* __restrict__ att,
                              float* __restrict__ logits, float* __restrict__ mx,
                              int E, int ET) {
  int t = blockIdx.x * blockDim.x + threadIdx.x;
  if (t >= ET * 8) return;
  int e = t >> 3, hh = t & 7;
  int s, d; float ea;
  if (e < E) { s = ei[e]; d = ei[E + e]; ea = eattr[e]; }
  else       { s = e - E; d = s;        ea = la[s]; }
  const float* xls = xl + (size_t)s * 128 + hh * 16;
  const float* xrd = xr + (size_t)d * 128 + hh * 16;
  const float* wep = We + hh * 16;
  const float* ap  = att + hh * 16;
  float acc = 0.f;
#pragma unroll
  for (int c = 0; c < 16; ++c) {
    float v = xls[c] + xrd[c] + ea * wep[c];
    acc += lrelu(v) * ap[c];
  }
  logits[(size_t)e * 8 + hh] = acc;
  atomicMaxF(&mx[(size_t)d * 8 + hh], acc);
}

// pexp = exp(logit - mx[dst]); den[dst][h] += pexp; num[dst][h*16+c] += pexp * xl[src][h*16+c]
__global__ void edge_scatter_k(const int* __restrict__ ei, const float* __restrict__ xl,
                               const float* __restrict__ logits, const float* __restrict__ mx,
                               float* __restrict__ den, float* __restrict__ num,
                               int E, int ET) {
  int t = blockIdx.x * blockDim.x + threadIdx.x;
  if (t >= ET * 8) return;
  int e = t >> 3, hh = t & 7;
  int s, d;
  if (e < E) { s = ei[e]; d = ei[E + e]; }
  else       { s = e - E; d = s; }
  float p = __expf(logits[(size_t)e * 8 + hh] - mx[(size_t)d * 8 + hh]);
  atomicAdd(&den[(size_t)d * 8 + hh], p);
  const float* xls = xl + (size_t)s * 128 + hh * 16;
  float* np = num + (size_t)d * 128 + hh * 16;
#pragma unroll
  for (int c = 0; c < 16; ++c) atomicAdd(&np[c], p * xls[c]);
}

// h = relu( bn(num/den + bias) + (resflag ? h : 0) )
__global__ void node_update_k(const float* __restrict__ num, const float* __restrict__ den,
                              const float* __restrict__ gbias,
                              const float* __restrict__ gamma, const float* __restrict__ beta,
                              const float* __restrict__ mean, const float* __restrict__ var,
                              float* __restrict__ h, int n, int resflag) {
  int t = blockIdx.x * blockDim.x + threadIdx.x;
  if (t >= n * 128) return;
  int c = t & 127, hh = c >> 4;
  int nd = t >> 7;
  float dd = fmaxf(den[(size_t)nd * 8 + hh], 1e-38f);
  float v = num[t] / dd + gbias[c];
  v = gamma[c] * (v - mean[c]) * rsqrtf(var[c] + 1e-5f) + beta[c];
  if (resflag) v += h[t];
  h[t] = relu_(v);
}

// ---------------- pooling + head ----------------
__global__ void pool_k(const float* __restrict__ h, const int* __restrict__ batch,
                       float* __restrict__ pooled, float* __restrict__ cnt, int n) {
  int t = blockIdx.x * blockDim.x + threadIdx.x;
  if (t >= n * 128) return;
  int nd = t >> 7, c = t & 127;
  int g = batch[nd];
  atomicAdd(&pooled[(size_t)g * 128 + c], h[t]);
  if (c == 0) atomicAdd(&cnt[g], 1.0f);
}

__global__ void head_k(const float* pooled, const float* cnt, const float* gf,
                       const float* mW1, const float* mb1, const float* mW2, const float* mb2,
                       const float* gW1, const float* gb1, const float* gW2, const float* gb2,
                       const float* pW1, const float* pb1, const float* pW2, const float* pb2,
                       const float* f1W, const float* f1b, const float* f2W, const float* f2b,
                       const float* f3W, const float* f3b, float* out) {
  __shared__ float z[192];
  __shared__ float t1[64];
  __shared__ float a1[128];
  __shared__ float a2[64];
  const int g = blockIdx.x;
  const int tid = threadIdx.x;  // 128 threads
  const float c = fmaxf(cnt[g], 1.0f);
  z[tid] = pooled[(size_t)g * 128 + tid] / c;
  const float* gfe = gf + (size_t)g * 16;
  if (tid < 16) {
    float a = mb1[tid];
    for (int i = 0; i < 4; ++i) a += gfe[i] * mW1[i * 16 + tid];
    t1[tid] = relu_(a);
  } else if (tid < 32) {
    int t = tid - 16; float a = gb1[t];
    for (int i = 0; i < 2; ++i) a += gfe[4 + i] * gW1[i * 16 + t];
    t1[16 + t] = relu_(a);
  } else if (tid < 64) {
    int t = tid - 32; float a = pb1[t];
    for (int i = 0; i < 10; ++i) a += gfe[6 + i] * pW1[i * 32 + t];
    t1[32 + t] = relu_(a);
  }
  __syncthreads();
  if (tid < 16) {
    float a = mb2[tid];
    for (int i = 0; i < 16; ++i) a += t1[i] * mW2[i * 16 + tid];
    z[128 + tid] = relu_(a);
  } else if (tid < 32) {
    int t = tid - 16; float a = gb2[t];
    for (int i = 0; i < 16; ++i) a += t1[16 + i] * gW2[i * 16 + t];
    z[144 + t] = relu_(a);
  } else if (tid < 64) {
    int t = tid - 32; float a = pb2[t];
    for (int i = 0; i < 32; ++i) a += t1[32 + i] * pW2[i * 32 + t];
    z[160 + t] = relu_(a);
  }
  __syncthreads();
  {
    float a = f1b[tid];
    for (int i = 0; i < 192; ++i) a += z[i] * f1W[i * 128 + tid];
    a1[tid] = relu_(a);
  }
  __syncthreads();
  if (tid < 64) {
    float a = f2b[tid];
    for (int i = 0; i < 128; ++i) a += a1[i] * f2W[i * 64 + tid];
    a2[tid] = relu_(a);
  }
  __syncthreads();
  if (tid == 0) {
    float a = f3b[0];
    for (int i = 0; i < 64; ++i) a += a2[i] * f3W[i];
    out[g] = a;
  }
}

// ---------------- host-side launch ----------------
static inline int cdiv(long long a, int b) { return (int)((a + b - 1) / b); }

extern "C" void kernel_launch(void* const* d_in, const int* in_sizes, int n_in,
                              void* d_out, int out_size, void* d_ws, size_t ws_size,
                              hipStream_t stream) {
  const int N = in_sizes[0] / 4;
  const int E = in_sizes[2];
  const int G = in_sizes[4] / 16;
  const int ET = E + N;

  const float* x     = (const float*)d_in[0];
  const int*   ei    = (const int*)d_in[1];
  const float* eattr = (const float*)d_in[2];
  const int*   batch = (const int*)d_in[3];
  const float* gf    = (const float*)d_in[4];
  const float* embW  = (const float*)d_in[5];
  const float* embB  = (const float*)d_in[6];

  // per-layer param tables: g = {Wl,bl,Wr,br,We,att,bias}, bn = {gamma,beta,mean,var}
  const int gbase[4]  = {7, 18, 29, 40};
  const int bnbase[4] = {14, 25, 36, 47};

  // workspace carve-out
  char* w = (char*)d_ws;
  auto alloc = [&](size_t bytes) -> void* {
    void* p = (void*)w;
    w += (bytes + 255) & ~(size_t)255;
    return p;
  };
  float* deg          = (float*)alloc((size_t)N * 4);
  float* la           = (float*)alloc((size_t)N * 4);
  unsigned short* hb  = (unsigned short*)alloc((size_t)N * 128 * 2);
  float* h            = (float*)alloc((size_t)N * 128 * 4);
  float* xl           = (float*)alloc((size_t)N * 128 * 4);
  float* xr           = (float*)alloc((size_t)N * 128 * 4);
  float* num          = (float*)alloc((size_t)N * 128 * 4);
  float* logits       = (float*)alloc((size_t)ET * 8 * 4);
  float* mx           = (float*)alloc((size_t)N * 8 * 4);
  float* den          = (float*)alloc((size_t)N * 8 * 4);
  unsigned short* wt  = (unsigned short*)alloc((size_t)256 * 128 * 2);
  float* pooled       = (float*)alloc((size_t)G * 128 * 4);
  float* cnt          = (float*)alloc((size_t)G * 4);
  (void)ws_size; (void)n_in;

  const int T = 256;

  // self-loop edge_attr (fill_value='mean')
  fillf_k<<<cdiv(N, T), T, 0, stream>>>(deg, 0.f, N);
  fillf_k<<<cdiv(N, T), T, 0, stream>>>(la, 0.f, N);
  deg_sum_k<<<cdiv(E, T), T, 0, stream>>>(ei, eattr, deg, la, E);
  loop_div_k<<<cdiv(N, T), T, 0, stream>>>(la, deg, N);

  // embedding
  embed_k<<<cdiv((long long)N * 64, T), T, 0, stream>>>(x, embW, embB, h, N);

  // 4 GATv2 layers
  for (int l = 0; l < 4; ++l) {
    const int din = (l == 0) ? 64 : 128;
    const float* Wl  = (const float*)d_in[gbase[l] + 0];
    const float* bl  = (const float*)d_in[gbase[l] + 1];
    const float* Wr  = (const float*)d_in[gbase[l] + 2];
    const float* br  = (const float*)d_in[gbase[l] + 3];
    const float* We  = (const float*)d_in[gbase[l] + 4];
    const float* att = (const float*)d_in[gbase[l] + 5];
    const float* gb  = (const float*)d_in[gbase[l] + 6];
    const float* bng = (const float*)d_in[bnbase[l] + 0];
    const float* bnb = (const float*)d_in[bnbase[l] + 1];
    const float* bnm = (const float*)d_in[bnbase[l] + 2];
    const float* bnv = (const float*)d_in[bnbase[l] + 3];

    cvt_h_k<<<cdiv((long long)N * din, T), T, 0, stream>>>(h, hb, N, din);
    prep_w_k<<<cdiv(256 * din, T), T, 0, stream>>>(Wl, Wr, wt, din);
    gemm_wmma_k<<<cdiv(N, 64), 512, 0, stream>>>(hb, wt, bl, br, xl, xr, N, din);

    fillf_k<<<cdiv((long long)N * 128, T), T, 0, stream>>>(num, 0.f, N * 128);
    fillf_k<<<cdiv((long long)N * 8, T), T, 0, stream>>>(mx, -INFINITY, N * 8);
    fillf_k<<<cdiv((long long)N * 8, T), T, 0, stream>>>(den, 0.f, N * 8);

    edge_logits_k<<<cdiv((long long)ET * 8, T), T, 0, stream>>>(
        ei, eattr, la, xl, xr, We, att, logits, mx, E, ET);
    edge_scatter_k<<<cdiv((long long)ET * 8, T), T, 0, stream>>>(
        ei, xl, logits, mx, den, num, E, ET);
    node_update_k<<<cdiv((long long)N * 128, T), T, 0, stream>>>(
        num, den, gb, bng, bnb, bnm, bnv, h, N, l > 0 ? 1 : 0);
  }

  // mean pooling per graph
  fillf_k<<<cdiv((long long)G * 128, T), T, 0, stream>>>(pooled, 0.f, G * 128);
  fillf_k<<<cdiv(G, T), T, 0, stream>>>(cnt, 0.f, G);
  pool_k<<<cdiv((long long)N * 128, T), T, 0, stream>>>(h, batch, pooled, cnt, N);

  // head MLPs -> out[G]
  head_k<<<G, 128, 0, stream>>>(
      pooled, cnt, gf,
      (const float*)d_in[51], (const float*)d_in[52], (const float*)d_in[53], (const float*)d_in[54],
      (const float*)d_in[55], (const float*)d_in[56], (const float*)d_in[57], (const float*)d_in[58],
      (const float*)d_in[59], (const float*)d_in[60], (const float*)d_in[61], (const float*)d_in[62],
      (const float*)d_in[63], (const float*)d_in[64], (const float*)d_in[65], (const float*)d_in[66],
      (const float*)d_in[67], (const float*)d_in[68], (float*)d_out);
  (void)out_size;
}
